// FilteredTensorProduct_25486335934481
// MI455X (gfx1250) — matrix operand order repaired and checked by reading, not compiled, not run on hardware
//
#include <hip/hip_runtime.h>

typedef _Float16 v16h __attribute__((ext_vector_type(16)));
typedef _Float16 v2h  __attribute__((ext_vector_type(2)));
typedef float    v8f  __attribute__((ext_vector_type(8)));
typedef float    v4f  __attribute__((ext_vector_type(4)));

// ---------------------------------------------------------------------------
// Setup: reconstruct W[a,b,c] = sum_p w_p * W3J_p[a,b,c]  (16x16x16 f32)
// ---------------------------------------------------------------------------

__device__ __forceinline__ double dfact(int n) {
  const double f[11] = {1.,1.,2.,6.,24.,120.,720.,5040.,40320.,362880.,3628800.};
  return f[n];
}

__device__ double su2_cg(int j1,int m1,int j2,int m2,int j3,int m3){
  if (m3 != m1 + m2) return 0.0;
  int vmin = -j1 + j2 + m3; if (-j1 + m1 > vmin) vmin = -j1 + m1; if (vmin < 0) vmin = 0;
  int vmax = j2 + j3 + m1;
  if (j3 - j1 + j2 < vmax) vmax = j3 - j1 + j2;
  if (j3 + m3     < vmax) vmax = j3 + m3;
  if (vmax < vmin) return 0.0;
  double C = (double)(2*j3+1)
    * dfact(j3+j1-j2)*dfact(j3-j1+j2)*dfact(j1+j2-j3)*dfact(j3+m3)*dfact(j3-m3)
    / (dfact(j1+j2+j3+1)*dfact(j1-m1)*dfact(j1+m1)*dfact(j2-m2)*dfact(j2+m2));
  double S = 0.0;
  for (int v = vmin; v <= vmax; ++v) {
    double t = dfact(j2+j3+m1-v)*dfact(j1-m1+v)
             / (dfact(v)*dfact(j3-j1+j2-v)*dfact(j3+m3-v)*dfact(v+j1-j2-m3));
    S += (((v + j2 + m2) & 1) ? -t : t);
  }
  return sqrt(C) * S;
}

struct cplx { double re, im; };
__device__ __forceinline__ cplx cmul(cplx a, cplx b){
  cplx r; r.re = a.re*b.re - a.im*b.im; r.im = a.re*b.im + a.im*b.re; return r;
}

// element [r][c] of (-i)^l * Q_l (real->complex change of basis)
__device__ cplx q_elem(int l, int r, int c){
  const double rt = 0.70710678118654752440;  // 1/sqrt(2)
  cplx q; q.re = 0.0; q.im = 0.0;
  int m = r - l;
  if (m < 0) {
    if      (c == l - m) q.re = rt;
    else if (c == l + m) q.im = -rt;
  } else if (m == 0) {
    if (c == l) q.re = 1.0;
  } else {
    double s = (m & 1) ? -1.0 : 1.0;
    if      (c == l + m) q.re = s * rt;
    else if (c == l - m) q.im = s * rt;
  }
  cplx o;
  switch (l & 3) {
    case 0:  o = q; break;
    case 1:  o.re =  q.im; o.im = -q.re; break;  // * (-i)
    case 2:  o.re = -q.re; o.im = -q.im; break;  // * (-1)
    default: o.re = -q.im; o.im =  q.re; break;  // * (+i)
  }
  return o;
}

__global__ void ftp_zero(float* __restrict__ w3) {
  for (int i = threadIdx.x; i < 16*16*16; i += blockDim.x) w3[i] = 0.0f;
}

// one block per path; path order matches the Python triple-nested enumeration
__global__ void ftp_paths(const float* __restrict__ weights, float* __restrict__ w3) {
  const int p = blockIdx.x;
  int lout = 0, l1 = 0, l2 = 0;
  {
    int cnt = 0;
    for (int lo = 0; lo <= 3; ++lo)
      for (int a = 0; a <= 3; ++a)
        for (int b = 0; b <= 3; ++b) {
          int lo_min = (a > b) ? (a - b) : (b - a);
          if ((((lo + a + b) & 1) == 0) && lo <= a + b && lo >= lo_min) {
            if (cnt == p) { lout = lo; l1 = a; l2 = b; }
            ++cnt;
          }
        }
  }
  const int d1 = 2*l1 + 1, d2 = 2*l2 + 1, d3 = 2*lout + 1;
  const int tot = d1 * d2 * d3;

  __shared__ double Cs[343];  // su2 CG tensor for this path (<=7x7x7)
  for (int idx = threadIdx.x; idx < tot; idx += blockDim.x) {
    int i = idx / (d2 * d3);
    int k = (idx / d3) % d2;
    int n = idx % d3;
    Cs[idx] = su2_cg(l1, i - l1, l2, k - l2, lout, n - lout);
  }
  __syncthreads();

  const double scale = sqrt((double)(2*lout + 1)) * (double)weights[p];
  for (int idx = threadIdx.x; idx < tot; idx += blockDim.x) {
    int j  = idx / (d2 * d3);
    int l  = (idx / d3) % d2;
    int mm = idx % d3;
    double acc = 0.0;
    for (int i = 0; i < d1; ++i) {
      cplx q1 = q_elem(l1, i, j);
      if (q1.re == 0.0 && q1.im == 0.0) continue;
      for (int k = 0; k < d2; ++k) {
        cplx q2 = q_elem(l2, k, l);
        if (q2.re == 0.0 && q2.im == 0.0) continue;
        cplx q12 = cmul(q1, q2);
        for (int n = 0; n < d3; ++n) {
          double cv = Cs[(i * d2 + k) * d3 + n];
          if (cv == 0.0) continue;
          cplx q3 = q_elem(lout, n, mm);
          // Re(q12 * conj(q3)) * cv
          acc += cv * (q12.re * q3.re + q12.im * q3.im);
        }
      }
    }
    int a = l1 * l1 + j;
    int b = l2 * l2 + l;
    int c = lout * lout + mm;
    w3[(a * 16 + b) * 16 + c] = (float)(scale * acc);  // path blocks are disjoint
  }
}

// pack W (256x16, K=ab major) into wave32 WMMA B-fragments, f16
// B layout (16-bit, 32x16): lanes 0-15 hold K 0..7,16..23, lanes 16-31 hold K 8..15,24..31,
// halves packed in K order within the v16h.
__global__ void ftp_pack(const float* __restrict__ w3, _Float16* __restrict__ bfrag) {
  const int tid  = threadIdx.x;      // 256 threads = 8 kb * 32 lanes
  const int kb   = tid >> 5;
  const int lane = tid & 31;
  const int n    = lane & 15;
  const int hi   = lane >> 4;
  for (int j = 0; j < 16; ++j) {
    int ab = kb * 32 + (j >> 3) * 16 + hi * 8 + (j & 7);
    bfrag[(kb * 32 + lane) * 16 + j] = (_Float16)w3[ab * 16 + n];
  }
}

// ---------------------------------------------------------------------------
// Main kernel: out(N x 16) = P(N x 256) * W(256 x 16), P built in registers.
// One wave handles 16 rows per tile; 8 x v_wmma_f32_16x16x32_f16 per tile.
// A fragment for lane (m,hi), k-block kb:
//   halves 0..7  = x1[m,2kb]   * x2[m, 8hi + j]
//   halves 8..15 = x1[m,2kb+1] * x2[m, 8hi + j]
// built with packed <2 x half> multiplies (v_pk_mul_f16).
// ---------------------------------------------------------------------------
__global__ __launch_bounds__(256) void ftp_main(
    const float* __restrict__ x1, const float* __restrict__ x2,
    const _Float16* __restrict__ bfrag, float* __restrict__ out, long ntiles)
{
  const int lane = (int)(threadIdx.x & 31u);
  const int hi   = lane >> 4;
  const int m    = lane & 15;
  const long wave   = (long)((blockIdx.x * blockDim.x + threadIdx.x) >> 5);
  const long nwaves = (long)((gridDim.x * blockDim.x) >> 5);

  // loop-invariant B fragments: 8 x v16h = 64 VGPRs, loaded once per wave
  v16h B[8];
#pragma unroll
  for (int kb = 0; kb < 8; ++kb)
    B[kb] = *(const v16h*)(bfrag + ((kb << 5) + lane) * 16);

  for (long t = wave; t < ntiles; t += nwaves) {
    const long row = t * 16 + m;
    const v4f* p1 = (const v4f*)(x1 + row * 16);
    const v4f* p2 = (const v4f*)(x2 + row * 16 + hi * 8);

    v4f xa0 = __builtin_nontemporal_load(p1 + 0);
    v4f xa1 = __builtin_nontemporal_load(p1 + 1);
    v4f xa2 = __builtin_nontemporal_load(p1 + 2);
    v4f xa3 = __builtin_nontemporal_load(p1 + 3);
    v4f xb0 = __builtin_nontemporal_load(p2 + 0);
    v4f xb1 = __builtin_nontemporal_load(p2 + 1);

    // x2 half-row as 4 packed half pairs
    v2h hp[4];
    hp[0] = (v2h){(_Float16)xb0[0], (_Float16)xb0[1]};
    hp[1] = (v2h){(_Float16)xb0[2], (_Float16)xb0[3]};
    hp[2] = (v2h){(_Float16)xb1[0], (_Float16)xb1[1]};
    hp[3] = (v2h){(_Float16)xb1[2], (_Float16)xb1[3]};

    // x1 row as 8 packed pairs: sp[kb] = { x1[m,2kb], x1[m,2kb+1] }
    v2h sp[8];
    sp[0] = (v2h){(_Float16)xa0[0], (_Float16)xa0[1]};
    sp[1] = (v2h){(_Float16)xa0[2], (_Float16)xa0[3]};
    sp[2] = (v2h){(_Float16)xa1[0], (_Float16)xa1[1]};
    sp[3] = (v2h){(_Float16)xa1[2], (_Float16)xa1[3]};
    sp[4] = (v2h){(_Float16)xa2[0], (_Float16)xa2[1]};
    sp[5] = (v2h){(_Float16)xa2[2], (_Float16)xa2[3]};
    sp[6] = (v2h){(_Float16)xa3[0], (_Float16)xa3[1]};
    sp[7] = (v2h){(_Float16)xa3[2], (_Float16)xa3[3]};

    v8f c = {};
#pragma unroll
    for (int kb = 0; kb < 8; ++kb) {
      const v2h s0v = (v2h){sp[kb].x, sp[kb].x};   // splat x1[m,2kb]   (op_sel)
      const v2h s1v = (v2h){sp[kb].y, sp[kb].y};   // splat x1[m,2kb+1]
      v16h a;
      v2h* ap = (v2h*)&a;
#pragma unroll
      for (int j = 0; j < 4; ++j) {
        ap[j]     = s0v * hp[j];                   // v_pk_mul_f16
        ap[4 + j] = s1v * hp[j];                   // v_pk_mul_f16
      }
      c = __builtin_amdgcn_wmma_f32_16x16x32_f16(
              false, a, false, B[kb], (short)0, c, false, false);
    }

    // C layout: lanes 0-15 -> M=v, N=lane; lanes 16-31 -> M=8+v, N=lane-16
    const long ro = t * 16 + (long)(hi * 8);
#pragma unroll
    for (int v = 0; v < 8; ++v)
      __builtin_nontemporal_store(c[v], out + (ro + v) * 16 + m);
  }
}

// scalar fallback for a (here empty) row remainder
__global__ void ftp_tail(const float* __restrict__ x1, const float* __restrict__ x2,
                         const float* __restrict__ w3, float* __restrict__ out,
                         long start, long n) {
  long r = start + blockIdx.x * blockDim.x + threadIdx.x;
  if (r >= n) return;
  float a[16], b[16], o[16];
  for (int i = 0; i < 16; ++i) { a[i] = x1[r * 16 + i]; b[i] = x2[r * 16 + i]; o[i] = 0.0f; }
  for (int i = 0; i < 16; ++i)
    for (int k = 0; k < 16; ++k) {
      float p = a[i] * b[k];
      const float* wr = w3 + (i * 16 + k) * 16;
      for (int cI = 0; cI < 16; ++cI) o[cI] += p * wr[cI];
    }
  for (int cI = 0; cI < 16; ++cI) out[r * 16 + cI] = o[cI];
}

extern "C" void kernel_launch(void* const* d_in, const int* in_sizes, int n_in,
                              void* d_out, int out_size, void* d_ws, size_t ws_size,
                              hipStream_t stream) {
  const float* x1 = (const float*)d_in[0];
  const float* x2 = (const float*)d_in[1];
  const float* w  = (const float*)d_in[2];
  float* out = (float*)d_out;

  const long n      = (long)in_sizes[0] / 16;   // rows
  const int  npath  = in_sizes[2];              // 23
  const long ntiles = n / 16;
  const long rem    = n - ntiles * 16;

  float*    w3    = (float*)d_ws;                          // 16 KB: W[a][b][c] f32
  _Float16* bfrag = (_Float16*)((char*)d_ws + 16384);      //  8 KB: B fragments f16

  ftp_zero<<<1, 256, 0, stream>>>(w3);
  ftp_paths<<<npath, 64, 0, stream>>>(w, w3);
  ftp_pack<<<1, 256, 0, stream>>>(w3, bfrag);

  if (ntiles > 0) {
    long want_blocks = (ntiles + 7) / 8;        // 8 waves per 256-thread block
    int blocks = (int)(want_blocks < 1024 ? want_blocks : 1024);
    if (blocks < 1) blocks = 1;
    ftp_main<<<blocks, 256, 0, stream>>>(x1, x2, bfrag, out, ntiles);
  }
  if (rem > 0) {
    ftp_tail<<<1, 32, 0, stream>>>(x1, x2, w3, out, ntiles * 16, n);
  }
}
